// TransformerBlock_8418135900662
// MI455X (gfx1250) — compile-verified, matrix operand order checked by reading, not run on hardware
//
#include <hip/hip_runtime.h>
#include <math.h>

// ---------------------------------------------------------------------------
// Transformer block on MI455X (gfx1250): all GEMMs via v_wmma_f32_16x16x32_bf16,
// GEMM tile staging via the Tensor Data Mover (tensor_load_to_lds) with a
// double-buffered pipeline (TDM streams tile i+1 while waves compute tile i).
// Fallback (no TDM builtin): global_load_b128 -> ds_store_b128.
//   S=4096, D=1024, H=16, E=64
// ---------------------------------------------------------------------------

typedef __attribute__((ext_vector_type(16))) __bf16 bf16x16;
typedef __attribute__((ext_vector_type(8)))  __bf16 bf16x8;
typedef __attribute__((ext_vector_type(8)))  float  f32x8;
typedef __attribute__((ext_vector_type(4)))  unsigned int u32x4;
typedef __attribute__((ext_vector_type(8)))  int i32x8;
typedef __attribute__((ext_vector_type(4)))  int i32x4;

#if defined(__has_builtin)
#  if __has_builtin(__builtin_amdgcn_tensor_load_to_lds)
#    define HAVE_TDM 1
#  endif
#endif

union BF16Frag {
    bf16x16 v;
    bf16x8  h[2];
};

__device__ __forceinline__ f32x8 wmma_bf16(bf16x16 a, bf16x16 b, f32x8 c) {
    // D = A(16x32 bf16) * B(32x16 bf16) + C(16x16 f32)
    return __builtin_amdgcn_wmma_f32_16x16x32_bf16(
        /*neg_a=*/false, a, /*neg_b=*/false, b,
        /*c_mod=*/(short)0, c, /*reuse_a=*/false, /*reuse_b=*/false);
}

__device__ __forceinline__ float gelu_tanh(float x) {
    float x3 = x * x * x;
    return 0.5f * x * (1.0f + tanhf(0.7978845608028654f * (x + 0.044715f * x3)));
}

template <int N>
__device__ __forceinline__ void wait_tensorcnt() {
#if defined(__has_builtin) && __has_builtin(__builtin_amdgcn_s_wait_tensorcnt)
    __builtin_amdgcn_s_wait_tensorcnt(N);
#else
    asm volatile("s_wait_tensorcnt %0" ::"i"(N) : "memory");
#endif
}

#ifdef HAVE_TDM
// Issue one TDM 2D tile load: global (rows x rowElems bf16, row stride
// strideElems) -> LDS at byte offset lds_off, with hardware row padding of
// 16B per 64B row (pad_interval=3 -> 64B, pad_amount=3 -> 4 DWORDs) so the
// LDS pitch is 80B = 40 elements, matching the WMMA fragment loads.
__device__ __forceinline__ void tdm_load_tile_2d(unsigned lds_off, const void* gptr,
                                                 unsigned rows, unsigned rowElems,
                                                 unsigned strideElems) {
    unsigned long long ga = (unsigned long long)gptr;
    unsigned td0 = strideElems;      // tensor dim0 (elements) - OOB bound only
    unsigned td1 = 0xFFFFu;          // tensor dim1 (rows)     - OOB bound only

    u32x4 g0 = {};
    g0[0] = 1u;                                        // count=1 (valid user D#)
    g0[1] = lds_off;                                   // lds_addr (bytes)
    g0[2] = (unsigned)(ga & 0xffffffffull);            // global_addr[31:0]
    g0[3] = (unsigned)((ga >> 32) & 0x1ffffffull)      // global_addr[56:32]
          | (2u << 30);                                // type = 2 ("image")

    i32x8 g1 = {};
    g1[0] = (int)((1u << 16)        // data_size = 1 -> 2 bytes
                | (1u << 20)        // pad_enable
                | (3u << 22)        // pad_interval: 16 DWORDs = 64B
                | (3u << 25));      // pad_amount:   4 DWORDs = 16B
    g1[1] = (int)((td0 & 0xffffu) << 16);                       // tensor_dim0 lo
    g1[2] = (int)(((td0 >> 16) & 0xffffu) | ((td1 & 0xffffu) << 16)); // dim0 hi | dim1 lo
    g1[3] = (int)(((td1 >> 16) & 0xffffu) | (rowElems << 16));  // dim1 hi | tile_dim0
    g1[4] = (int)(rows & 0xffffu);                              // tile_dim1 (tile_dim2=0)
    g1[5] = (int)strideElems;                                   // tensor_dim0_stride lo32
    g1[6] = 0;                                                  // stride hi16 | dim1_stride lo
    g1[7] = 0;

    i32x4 g2 = {};
    i32x4 g3 = {};
#if __clang_major__ >= 23
    i32x8 g4 = {};
    __builtin_amdgcn_tensor_load_to_lds(g0, g1, g2, g3, g4, 0);
#else
    __builtin_amdgcn_tensor_load_to_lds(g0, g1, g2, g3, 0);
#endif
}
#endif  // HAVE_TDM

// ------------------------- conversion / transpose kernels -------------------

__global__ __launch_bounds__(256) void cvt_f32_bf16(const float* __restrict__ in,
                                                    __bf16* __restrict__ out, int n) {
    int i = blockIdx.x * 256 + threadIdx.x;
    if (i < n) out[i] = (__bf16)in[i];
}

// in: f32 [K_, N_] row-major ; out: bf16 [N_, K_] row-major  (i.e. B^T for NT gemm)
__global__ __launch_bounds__(256) void transpose_f32_to_bf16(const float* __restrict__ in,
                                                             __bf16* __restrict__ out,
                                                             int K_, int N_) {
    int idx = blockIdx.x * 256 + threadIdx.x;
    if (idx >= K_ * N_) return;
    int k = idx / N_, n = idx % N_;
    out[(size_t)n * K_ + k] = (__bf16)in[idx];
}

// in: f32 [H, D, E] ; out slot w of bf16 [H, 3, E, D]
__global__ __launch_bounds__(256) void transpose_qkv(const float* __restrict__ in,
                                                     __bf16* __restrict__ out, int w) {
    int idx = blockIdx.x * 256 + threadIdx.x;       // H*D*E = 1M
    if (idx >= 16 * 1024 * 64) return;
    int h = idx / (1024 * 64);
    int r = idx % (1024 * 64);
    int d = r / 64;
    int e = r % 64;
    out[(((size_t)h * 3 + w) * 64 + e) * 1024 + d] = (__bf16)in[idx];
}

// ------------------------------- NT GEMM ------------------------------------
// C[M,N] = A[M,K] * B[N,K]^T  (A,B bf16 row-major over K), f32 accumulate.
// Block: 256 threads = 8 waves (4 along M x 2 along N). Block tile 128x128,
// wave tile 32x64 (2x4 accumulators), K-step 32. Double-buffered TDM staging:
// tensor ops complete in order per wave, so after issuing tile i+1's two
// loads, s_wait_tensorcnt<=2 guarantees tile i has fully landed.

__global__ __launch_bounds__(256) void gemm_nt_bf16(
    const __bf16* __restrict__ A, const __bf16* __restrict__ Bm,
    int M, int N, int K, int mode,
    const float* __restrict__ bias,    // mode 1/2/3
    const float* __restrict__ resid,   // mode 1: x ; mode 3: y
    __bf16* __restrict__ outKb,        // mode 0
    __bf16* __restrict__ outQb,        // mode 0
    __bf16* __restrict__ outVt,        // mode 0
    float*  __restrict__ outF,         // mode 1: y ; mode 3: final out
    __bf16* __restrict__ outBf)        // mode 1: y bf16 ; mode 2: h bf16
{
    constexpr int BM = 128, BN = 128, BK = 32, LP = 40;  // LP: LDS row pitch (elems)
    __shared__ __attribute__((aligned(16))) __bf16 As[2][BM * LP];
    __shared__ __attribute__((aligned(16))) __bf16 Bs[2][BN * LP];

    const int tid   = threadIdx.x;
    const int wave  = tid >> 5;
    const int lane  = tid & 31;
    const int rl    = lane & 15;
    const int hi    = lane >> 4;
    const int waveM = wave & 3;        // 0..3
    const int waveN = wave >> 2;       // 0..1
    const int m0    = blockIdx.y * BM;
    const int n0    = blockIdx.x * BN;

    f32x8 acc[2][4] = {};

    const int nk = K / BK;

#ifdef HAVE_TDM
    if (wave == 0) {
        tdm_load_tile_2d((unsigned)(unsigned long long)(uintptr_t)&As[0][0],
                         A + (size_t)m0 * K, BM, BK, (unsigned)K);
        tdm_load_tile_2d((unsigned)(unsigned long long)(uintptr_t)&Bs[0][0],
                         Bm + (size_t)n0 * K, BN, BK, (unsigned)K);
    }
#endif

    for (int it = 0; it < nk; ++it) {
        const int cur = it & 1;
#ifdef HAVE_TDM
        if (wave == 0) {
            if (it + 1 < nk) {
                const int nxt = (it + 1) & 1;
                const int k1  = (it + 1) * BK;
                tdm_load_tile_2d((unsigned)(unsigned long long)(uintptr_t)&As[nxt][0],
                                 A + (size_t)m0 * K + k1, BM, BK, (unsigned)K);
                tdm_load_tile_2d((unsigned)(unsigned long long)(uintptr_t)&Bs[nxt][0],
                                 Bm + (size_t)n0 * K + k1, BN, BK, (unsigned)K);
                wait_tensorcnt<2>();   // tile `it` (two older ops) has landed
            } else {
                wait_tensorcnt<0>();   // final tile
            }
        }
        __syncthreads();
#else
        // ---- manual staging: A tile 128x32, B tile 128x32 (b128 each) ----
        {
            const int k0 = it * BK;
#pragma unroll
            for (int c = 0; c < 2; ++c) {
                int chunk = tid + c * 256;              // 0..511
                int row = chunk >> 2;
                int col = (chunk & 3) << 3;
                bf16x8 v = *(const bf16x8*)(A + (size_t)(m0 + row) * K + k0 + col);
                *(bf16x8*)(As[cur] + row * LP + col) = v;
            }
#pragma unroll
            for (int c = 0; c < 2; ++c) {
                int chunk = tid + c * 256;
                int row = chunk >> 2;
                int col = (chunk & 3) << 3;
                bf16x8 v = *(const bf16x8*)(Bm + (size_t)(n0 + row) * K + k0 + col);
                *(bf16x8*)(Bs[cur] + row * LP + col) = v;
            }
        }
        __syncthreads();
#endif

        // ---- fragments per documented 16-bit A/B layout ----
        BF16Frag afrag[2], bfrag[4];
#pragma unroll
        for (int mi = 0; mi < 2; ++mi) {
            const __bf16* p = As[cur] + (waveM * 32 + mi * 16 + rl) * LP;
            afrag[mi].h[0] = *(const bf16x8*)(p + hi * 8);
            afrag[mi].h[1] = *(const bf16x8*)(p + 16 + hi * 8);
        }
#pragma unroll
        for (int ni = 0; ni < 4; ++ni) {
            const __bf16* p = Bs[cur] + (waveN * 64 + ni * 16 + rl) * LP;
            bfrag[ni].h[0] = *(const bf16x8*)(p + hi * 8);
            bfrag[ni].h[1] = *(const bf16x8*)(p + 16 + hi * 8);
        }
#pragma unroll
        for (int mi = 0; mi < 2; ++mi)
#pragma unroll
            for (int ni = 0; ni < 4; ++ni)
                acc[mi][ni] = wmma_bf16(afrag[mi].v, bfrag[ni].v, acc[mi][ni]);
        __syncthreads();   // buffer `cur` drained before tile it+2 targets it
    }

    // ---- epilogue (C layout: vgpr i -> row hi*8+i, lane -> col rl) ----
#pragma unroll
    for (int mi = 0; mi < 2; ++mi)
#pragma unroll
        for (int ni = 0; ni < 4; ++ni)
#pragma unroll
            for (int i = 0; i < 8; ++i) {
                int m = m0 + waveM * 32 + mi * 16 + hi * 8 + i;
                int n = n0 + waveN * 64 + ni * 16 + rl;
                float v = acc[mi][ni][i];
                if (mode == 0) {
                    // n in [0,3072): h = n/192, slot = (n%192)/64 (0=K,1=Q,2=V), e = n%64
                    int h = n / 192, r = n % 192, w = r / 64, e = r % 64;
                    if (w == 0)
                        outKb[((size_t)h * 4096 + m) * 64 + e] = (__bf16)v;
                    else if (w == 1)
                        outQb[((size_t)h * 4096 + m) * 64 + e] = (__bf16)v;
                    else
                        outVt[((size_t)h * 64 + e) * 4096 + m] = (__bf16)v;   // V^T
                } else if (mode == 1) {
                    float y = v + bias[n] + resid[(size_t)m * N + n];
                    outF[(size_t)m * N + n]  = y;
                    outBf[(size_t)m * N + n] = (__bf16)y;
                } else if (mode == 2) {
                    outBf[(size_t)m * N + n] = (__bf16)gelu_tanh(v + bias[n]);
                } else {
                    outF[(size_t)m * N + n] = v + bias[n] + resid[(size_t)m * N + n];
                }
            }
}

// ------------------------- reversed flash attention -------------------------
// scores[h,s,t] = sum_e K[h,s,e]*Q[h,t,e] / 8, mask t<=s, softmax over t,
// out[h,s,e] = sum_t w[h,s,t] * V[h,t,e].  One wave per (h, 16-row s block).
// Online softmax over t-tiles of 32; P re-laid out C->A via per-wave LDS.

__global__ __launch_bounds__(128) void flash_attn_rev(
    const __bf16* __restrict__ Kb,   // [H,S,E]
    const __bf16* __restrict__ Qb,   // [H,S,E]
    const __bf16* __restrict__ Vt,   // [H,E,S]  (V transposed)
    __bf16* __restrict__ attnOut)    // [S, H*E]
{
    constexpr int S = 4096, E = 64, PLD = 40;
    __shared__ __attribute__((aligned(16))) __bf16 Ps[4][16 * PLD];

    const int wave = threadIdx.x >> 5;
    const int lane = threadIdx.x & 31;
    const int rl   = lane & 15;
    const int hi   = lane >> 4;
    const int gw   = blockIdx.x * 4 + wave;
    const int h    = gw >> 8;          // 256 s-blocks per head
    const int s0   = (gw & 255) * 16;

    const __bf16* Kh = Kb + (size_t)h * S * E;
    const __bf16* Qh = Qb + (size_t)h * S * E;
    const __bf16* Vh = Vt + (size_t)h * E * S;

    // K block fragments (A matrix, rows s0..s0+15, E split into two K=32 halves)
    BF16Frag kfrag[2];
#pragma unroll
    for (int eh = 0; eh < 2; ++eh) {
        const __bf16* p = Kh + (size_t)(s0 + rl) * E + eh * 32;
        kfrag[eh].h[0] = *(const bf16x8*)(p + hi * 8);
        kfrag[eh].h[1] = *(const bf16x8*)(p + 16 + hi * 8);
    }

    f32x8 oacc[4] = {};
    float mrow[8], lrow[8];
#pragma unroll
    for (int i = 0; i < 8; ++i) { mrow[i] = -1e30f; lrow[i] = 0.0f; }

    const int tEnd = s0 + 16;               // need t <= s, s < s0+16
    for (int t0 = 0; t0 < tEnd; t0 += 32) {
        // prefetch next t-tile (lowers to global_prefetch_b8)
        if (t0 + 32 < tEnd) {
            __builtin_prefetch(Qh + (size_t)(t0 + 32 + rl) * E, 0, 0);
            __builtin_prefetch(Vh + (size_t)rl * S + t0 + 32, 0, 0);
        }
        // ---- scores: two 16x16 tiles (columns t0..t0+15, t0+16..t0+31) ----
        f32x8 sacc[2] = {};
#pragma unroll
        for (int tt = 0; tt < 2; ++tt) {
            const __bf16* p = Qh + (size_t)(t0 + tt * 16 + rl) * E;
            BF16Frag qf[2];
#pragma unroll
            for (int eh = 0; eh < 2; ++eh) {
                qf[eh].h[0] = *(const bf16x8*)(p + eh * 32 + hi * 8);
                qf[eh].h[1] = *(const bf16x8*)(p + eh * 32 + 16 + hi * 8);
            }
            sacc[tt] = wmma_bf16(kfrag[0].v, qf[0].v, sacc[tt]);
            sacc[tt] = wmma_bf16(kfrag[1].v, qf[1].v, sacc[tt]);
        }
        // ---- scale + causal mask (t <= s) ----
#pragma unroll
        for (int tt = 0; tt < 2; ++tt)
#pragma unroll
            for (int i = 0; i < 8; ++i) {
                int t = t0 + tt * 16 + rl;
                int s = s0 + hi * 8 + i;
                float v = sacc[tt][i] * 0.125f;      // 1/sqrt(64)
                sacc[tt][i] = (t <= s) ? v : -1e30f;
            }
        // ---- online softmax per row (stats duplicated across 16-lane half) ----
#pragma unroll
        for (int i = 0; i < 8; ++i) {
            float tm = fmaxf(sacc[0][i], sacc[1][i]);
#pragma unroll
            for (int d = 1; d < 16; d <<= 1) tm = fmaxf(tm, __shfl_xor(tm, d, 32));
            float mnew = fmaxf(mrow[i], tm);
            float corr = __expf(mrow[i] - mnew);
            mrow[i] = mnew;
            float p0 = __expf(sacc[0][i] - mnew);
            float p1 = __expf(sacc[1][i] - mnew);
            float rs = p0 + p1;
#pragma unroll
            for (int d = 1; d < 16; d <<= 1) rs += __shfl_xor(rs, d, 32);
            lrow[i] = lrow[i] * corr + rs;
#pragma unroll
            for (int et = 0; et < 4; ++et) oacc[et][i] *= corr;
            // P tile to LDS in row-major (rows s-local, cols t-local)
            Ps[wave][(hi * 8 + i) * PLD + rl]      = (__bf16)p0;
            Ps[wave][(hi * 8 + i) * PLD + 16 + rl] = (__bf16)p1;
        }
        // ---- reload P as A fragment (same-wave DS ops are in order) ----
        BF16Frag pf;
        {
            const __bf16* pp = &Ps[wave][rl * PLD];
            pf.h[0] = *(const bf16x8*)(pp + hi * 8);
            pf.h[1] = *(const bf16x8*)(pp + 16 + hi * 8);
        }
        // ---- oacc += P(16x32) * V(32x64) via V^T fragments ----
#pragma unroll
        for (int et = 0; et < 4; ++et) {
            const __bf16* vp = Vh + (size_t)(et * 16 + rl) * S + t0;
            BF16Frag vf;
            vf.h[0] = *(const bf16x8*)(vp + hi * 8);
            vf.h[1] = *(const bf16x8*)(vp + 16 + hi * 8);
            oacc[et] = wmma_bf16(pf.v, vf.v, oacc[et]);
        }
    }

    // ---- normalize and write concat-head output [S, H*E] ----
#pragma unroll
    for (int et = 0; et < 4; ++et)
#pragma unroll
        for (int i = 0; i < 8; ++i) {
            int s = s0 + hi * 8 + i;
            int f = h * 64 + et * 16 + rl;
            attnOut[(size_t)s * 1024 + f] = (__bf16)(oacc[et][i] / lrow[i]);
        }
}

// ------------------------------- launcher -----------------------------------

extern "C" void kernel_launch(void* const* d_in, const int* in_sizes, int n_in,
                              void* d_out, int out_size, void* d_ws, size_t ws_size,
                              hipStream_t stream) {
    (void)in_sizes; (void)n_in; (void)out_size; (void)ws_size;

    constexpr int S = 4096, D = 1024, H = 16, E = 64;

    const float* x  = (const float*)d_in[0];
    const float* Wk = (const float*)d_in[1];
    const float* Wq = (const float*)d_in[2];
    const float* Wv = (const float*)d_in[3];
    const float* Wo = (const float*)d_in[4];
    const float* bo = (const float*)d_in[5];
    const float* W1 = (const float*)d_in[6];
    const float* b1 = (const float*)d_in[7];
    const float* W2 = (const float*)d_in[8];
    const float* b2 = (const float*)d_in[9];
    float* out = (float*)d_out;

    // workspace carve-out (~126 MB total)
    char*  ws  = (char*)d_ws;
    size_t off = 0;
    auto alloc = [&](size_t bytes) {
        void* p = ws + off;
        off = (off + bytes + 255) & ~(size_t)255;
        return p;
    };
    __bf16* Xbf   = (__bf16*)alloc((size_t)S * D * 2);           // 8 MB
    __bf16* Wqkvt = (__bf16*)alloc((size_t)H * 3 * E * D * 2);   // 6 MB  [H,3,E,D]
    __bf16* Wot   = (__bf16*)alloc((size_t)D * H * E * 2);       // 2 MB  [D, H*E]
    __bf16* W1t   = (__bf16*)alloc((size_t)4 * D * D * 2);       // 8 MB  [4D, D]
    __bf16* W2t   = (__bf16*)alloc((size_t)D * 4 * D * 2);       // 8 MB  [D, 4D]
    __bf16* Kbuf  = (__bf16*)alloc((size_t)H * S * E * 2);       // 8 MB
    __bf16* Qbuf  = (__bf16*)alloc((size_t)H * S * E * 2);       // 8 MB
    __bf16* Vtb   = (__bf16*)alloc((size_t)H * E * S * 2);       // 8 MB
    __bf16* attnb = (__bf16*)alloc((size_t)S * H * E * 2);       // 8 MB
    float*  yf    = (float*) alloc((size_t)S * D * 4);           // 16 MB
    __bf16* ybf   = (__bf16*)alloc((size_t)S * D * 2);           // 8 MB
    __bf16* hbf   = (__bf16*)alloc((size_t)S * 4 * D * 2);       // 32 MB

    // 1) conversions / weight transposes (bf16, NT layout)
    cvt_f32_bf16<<<(S * D) / 256, 256, 0, stream>>>(x, Xbf, S * D);
    transpose_qkv<<<(H * D * E) / 256, 256, 0, stream>>>(Wk, Wqkvt, 0);
    transpose_qkv<<<(H * D * E) / 256, 256, 0, stream>>>(Wq, Wqkvt, 1);
    transpose_qkv<<<(H * D * E) / 256, 256, 0, stream>>>(Wv, Wqkvt, 2);
    transpose_f32_to_bf16<<<(H * E * D) / 256, 256, 0, stream>>>(Wo, Wot, H * E, D);
    transpose_f32_to_bf16<<<(D * 4 * D) / 256, 256, 0, stream>>>(W1, W1t, D, 4 * D);
    transpose_f32_to_bf16<<<(4 * D * D) / 256, 256, 0, stream>>>(W2, W2t, 4 * D, D);

    // 2) fused QKV projection: N = H*3*E = 3072
    gemm_nt_bf16<<<dim3(3072 / 128, S / 128), 256, 0, stream>>>(
        Xbf, Wqkvt, S, 3072, D, /*mode=*/0,
        nullptr, nullptr, Kbuf, Qbuf, Vtb, nullptr, nullptr);

    // 3) reversed-causal attention -> attnb [S, H*E]
    flash_attn_rev<<<(H * (S / 16)) / 4, 128, 0, stream>>>(Kbuf, Qbuf, Vtb, attnb);

    // 4) output projection + bias + residual -> y (f32 + bf16)
    gemm_nt_bf16<<<dim3(D / 128, S / 128), 256, 0, stream>>>(
        attnb, Wot, S, D, H * E, /*mode=*/1,
        bo, x, nullptr, nullptr, nullptr, yf, ybf);

    // 5) MLP up + gelu -> h (bf16)
    gemm_nt_bf16<<<dim3(4 * D / 128, S / 128), 256, 0, stream>>>(
        ybf, W1t, S, 4 * D, D, /*mode=*/2,
        b1, nullptr, nullptr, nullptr, nullptr, nullptr, hbf);

    // 6) MLP down + bias + residual -> final output (f32)
    gemm_nt_bf16<<<dim3(D / 128, S / 128), 256, 0, stream>>>(
        hbf, W2t, S, D, 4 * D, /*mode=*/3,
        b2, yf, nullptr, nullptr, nullptr, out, nullptr);
}
